// Multihead_Attention_46806553592368
// MI455X (gfx1250) — matrix operand order repaired and checked by reading, not compiled, not run on hardware
//
#include <hip/hip_runtime.h>
#include <hip/hip_bf16.h>
#include <stdint.h>

// ===========================================================================
// Multihead attention for MI455X (gfx1250, wave32, WMMA).
//
// Shapes: B=4, S=2048, E=1024, h=16, head_dim=64.
// All GEMMs are M=8192, N=1024, K=1024.
//
// Faithful to reference quirks:
//  * reshape(h,B,S,64) of contiguous (B,S,1024)  -> pure reinterpretation,
//    slice (h,b) lives at flat offset (h*4+b)*2048*64.
//  * softmax over dim=1 (the batch axis, size 4) -> elementwise 4-way
//    softmax across batch score tiles; no materialized 2048x2048 scores.
//
// Precision: bf16 WMMA (v_wmma_f32_16x16x32_bf16), fp32 accumulate.
// Compute-bound problem (~137 GFLOP vs ~100MB traffic, L2-resident).
// ===========================================================================

#if defined(__has_builtin)
#  if __has_builtin(__builtin_amdgcn_cvt_pk_bf16_f32)
#    define USE_CVT_PK_BF16 1
#  endif
#  if __has_builtin(__builtin_amdgcn_global_load_async_to_lds_b128)
#    define USE_ASYNC_LDS 1
#  endif
#  if __has_builtin(__builtin_amdgcn_s_wait_asynccnt)
#    define WAIT_ASYNC() __builtin_amdgcn_s_wait_asynccnt(0)
#  else
#    define WAIT_ASYNC() asm volatile("s_wait_asynccnt 0x0" ::: "memory")
#  endif
#else
#  define WAIT_ASYNC() asm volatile("s_wait_asynccnt 0x0" ::: "memory")
#endif

typedef __attribute__((ext_vector_type(16))) __bf16 v16bf;
typedef __attribute__((ext_vector_type(2)))  __bf16 v2bf;
typedef __attribute__((ext_vector_type(8)))  float  v8f;
typedef __attribute__((ext_vector_type(4)))  int    v4i;
typedef __attribute__((address_space(1)))    v4i    gv4i;   // global int4
typedef __attribute__((address_space(3)))    v4i    lv4i;   // LDS int4

union BF16Frag { uint4 u4[2]; v16bf v; };
union F32x8    { v8f v; float f[8]; float4 f4[2]; };

// fp32 -> 2x bf16 packed. Native __bf16 cast lets the backend pick the
// hardware converter (v_cvt_pk_bf16_f32 / v_cvt_bf16_f32) when present.
__device__ __forceinline__ unsigned int pack2bf(float lo, float hi) {
#ifdef USE_CVT_PK_BF16
  auto v = __builtin_amdgcn_cvt_pk_bf16_f32(lo, hi);
  return __builtin_bit_cast(unsigned int, v);
#else
  v2bf v;
  v.x = (__bf16)lo;
  v.y = (__bf16)hi;
  return __builtin_bit_cast(unsigned int, v);
#endif
}
__device__ __forceinline__ unsigned short f2bf(float f) {
  __bf16 b = (__bf16)f;
  return __builtin_bit_cast(unsigned short, b);
}

#define GM 8192
#define GN 1024
#define GK 1024
#define KT 32          // WMMA K per step
#define LDP 40         // LDS row pitch (u16): 80B, bank-friendly, 16B aligned

// ---------------------------------------------------------------------------
// GEMM core: C(M=8192,N=1024) = A(8192x1024) @ W^T(1024x1024) + bias
//   A row-major (fp32 or bf16), W row-major (N x K) fp32.
//   out_mode 0: bf16 row-major     (Q, K projections)
//   out_mode 1: bf16 "Vt" layout   (V projection, stored [(h*4+b)*64+d][key])
//   out_mode 2: fp32 row-major     (final output projection)
// Block = 256 threads (8 waves, 2x4), tile 128(M) x 256(N), 64x64 per wave.
// ---------------------------------------------------------------------------
template <bool A_BF16>
__device__ __forceinline__ void gemm_core(const void* __restrict__ Aptr,
                                          const float* __restrict__ W,
                                          const float* __restrict__ bias,
                                          void* __restrict__ Cptr,
                                          int out_mode) {
  __shared__ unsigned short As[128 * LDP];   // 10 KB
  __shared__ unsigned short Ws[256 * LDP];   // 20 KB

  const int tid  = threadIdx.x;
  const int lane = tid & 31;
  const int wave = tid >> 5;
  const int wm   = (wave >> 2) * 64;   // wave M origin inside 128
  const int wn   = (wave & 3) * 64;    // wave N origin inside 256
  const int m0   = blockIdx.y * 128;
  const int n0   = blockIdx.x * 256;
  const int l15  = lane & 15;
  const int lh   = lane >> 4;          // half-wave select

  F32x8 acc[4][4];
#pragma unroll
  for (int mi = 0; mi < 4; ++mi)
#pragma unroll
    for (int ni = 0; ni < 4; ++ni)
#pragma unroll
      for (int r = 0; r < 8; ++r) acc[mi][ni].f[r] = 0.0f;

  for (int ks = 0; ks < GK; ks += KT) {
    __syncthreads();
    // ---- stage A tile (128 x 32) into LDS as bf16 ----
    if (A_BF16) {
      const unsigned short* Ab = (const unsigned short*)Aptr;
#ifdef USE_ASYNC_LDS
      // Pure copy: use the gfx1250 async global->LDS DMA path (ASYNCcnt).
#pragma unroll
      for (int i = 0; i < 2; ++i) {            // 512 x b128 / 256 threads
        int li = tid + i * 256;
        int r = li >> 2, c = li & 3;
        const unsigned short* gp = Ab + (size_t)(m0 + r) * GK + ks + c * 8;
        __builtin_amdgcn_global_load_async_to_lds_b128(
            (gv4i*)gp, (lv4i*)&As[r * LDP + c * 8], 0, 0);
      }
      WAIT_ASYNC();
#else
#pragma unroll
      for (int i = 0; i < 2; ++i) {
        int li = tid + i * 256;
        int r = li >> 2, c = li & 3;           // 4 x (8 bf16) per row
        uint4 v = *(const uint4*)(Ab + (size_t)(m0 + r) * GK + ks + c * 8);
        *(uint4*)(&As[r * LDP + c * 8]) = v;
      }
#endif
    } else {
      const float* Af = (const float*)Aptr;
#pragma unroll
      for (int i = 0; i < 4; ++i) {            // 1024 float4 / 256 threads
        int li = tid + i * 256;
        int r = li >> 3, c = li & 7;           // 8 x float4 per row
        float4 v = *(const float4*)(Af + (size_t)(m0 + r) * GK + ks + c * 4);
        uint2 p; p.x = pack2bf(v.x, v.y); p.y = pack2bf(v.z, v.w);
        *(uint2*)(&As[r * LDP + c * 4]) = p;
        if (ks + KT < GK)                      // global_prefetch_b8 next K-slab
          __builtin_prefetch((const void*)(Af + (size_t)(m0 + r) * GK + ks + KT + c * 4), 0, 1);
      }
    }
    // ---- stage W tile (256 x 32) into LDS as bf16 ----
#pragma unroll
    for (int i = 0; i < 8; ++i) {              // 2048 float4 / 256 threads
      int li = tid + i * 256;
      int r = li >> 3, c = li & 7;
      float4 v = *(const float4*)(W + (size_t)(n0 + r) * GK + ks + c * 4);
      uint2 p; p.x = pack2bf(v.x, v.y); p.y = pack2bf(v.z, v.w);
      *(uint2*)(&Ws[r * LDP + c * 4]) = p;
    }
    __syncthreads();

    // ---- fragments ----
    // A 16x32 bf16: lanes 0-15 row M=l, K {0..7,16..23}; lanes 16-31 K {8..15,24..31}
    BF16Frag a[4];
#pragma unroll
    for (int mi = 0; mi < 4; ++mi) {
      const unsigned short* p = &As[(wm + mi * 16 + l15) * LDP + lh * 8];
      a[mi].u4[0] = *(const uint4*)(p);
      a[mi].u4[1] = *(const uint4*)(p + 16);
    }
    // B 32x16 bf16: lane col N=l&15; lanes 0-15 K {0..7,16..23}, lanes 16-31 K {8..15,24..31}
    BF16Frag bfrag[4];
#pragma unroll
    for (int ni = 0; ni < 4; ++ni) {
      const unsigned short* p = &Ws[(wn + ni * 16 + l15) * LDP + lh * 8];
      bfrag[ni].u4[0] = *(const uint4*)(p);
      bfrag[ni].u4[1] = *(const uint4*)(p + 16);
    }
#pragma unroll
    for (int mi = 0; mi < 4; ++mi)
#pragma unroll
      for (int ni = 0; ni < 4; ++ni)
        acc[mi][ni].v = __builtin_amdgcn_wmma_f32_16x16x32_bf16(
            false, a[mi].v, false, bfrag[ni].v, (short)0, acc[mi][ni].v,
            false, false);
  }

  // ---- epilogue: C/D layout => row m = r + 8*lh, col n = l15 ----
#pragma unroll
  for (int mi = 0; mi < 4; ++mi)
#pragma unroll
    for (int ni = 0; ni < 4; ++ni) {
      const int col   = n0 + wn + ni * 16 + l15;
      const float bv  = bias[col];
      const int mbase = m0 + wm + mi * 16 + lh * 8;
      if (out_mode == 2) {
        float* C = (float*)Cptr;
#pragma unroll
        for (int r = 0; r < 8; ++r)
          C[(size_t)(mbase + r) * GN + col] = acc[mi][ni].f[r] + bv;
      } else if (out_mode == 0) {
        unsigned short* C = (unsigned short*)Cptr;
#pragma unroll
        for (int r = 0; r < 8; ++r)
          C[(size_t)(mbase + r) * GN + col] = f2bf(acc[mi][ni].f[r] + bv);
      } else {  // V transposed: flat=(row*1024+col); [(h*4+b)*64+d][key]
        unsigned short* C = (unsigned short*)Cptr;
#pragma unroll
        for (int r = 0; r < 8; ++r) {
          unsigned int flat = (unsigned int)(mbase + r) * 1024u + (unsigned int)col;
          unsigned int hb  = flat >> 17;
          unsigned int key = (flat >> 6) & 2047u;
          unsigned int d   = flat & 63u;
          C[((size_t)hb * 64 + d) * 2048 + key] = f2bf(acc[mi][ni].f[r] + bv);
        }
      }
    }
}

__global__ __launch_bounds__(256) void qkv_proj_kernel(
    const float* __restrict__ x,  const float* __restrict__ y,
    const float* __restrict__ Wq, const float* __restrict__ bq,
    const float* __restrict__ Wk, const float* __restrict__ bk,
    const float* __restrict__ Wv, const float* __restrict__ bv,
    unsigned short* __restrict__ Qb, unsigned short* __restrict__ Kb,
    unsigned short* __restrict__ Vt) {
  const int z = blockIdx.z;
  const float* A = (z == 0) ? x : y;
  const float* W = (z == 0) ? Wq : (z == 1) ? Wk : Wv;
  const float* b = (z == 0) ? bq : (z == 1) ? bk : bv;
  void* C = (z == 0) ? (void*)Qb : (z == 1) ? (void*)Kb : (void*)Vt;
  gemm_core<false>(A, W, b, C, (z == 2) ? 1 : 0);
}

__global__ __launch_bounds__(256) void out_proj_kernel(
    const unsigned short* __restrict__ Ob, const float* __restrict__ Wo,
    const float* __restrict__ bo, float* __restrict__ out) {
  gemm_core<true>(Ob, Wo, bo, out, 2);
}

// ---------------------------------------------------------------------------
// Attention kernel.
// Grid (h=16, qtiles=32), block = 128 threads = 4 waves; wave w owns batch b=w.
// Per k-tile of 32 keys:
//   S_b(64x32) = Q_hb @ K_hb^T   (WMMA, contraction over d=64)
//   scores -> LDS, barrier, elementwise softmax over the 4 batches,
//   attn bf16 -> LDS (wave-private section, no barrier needed),
//   O_b(64x64) += attn_b @ V_b   (WMMA, B-fragments direct from Vt global)
// ---------------------------------------------------------------------------
#define S_LEN 2048
#define HD 64

__global__ __launch_bounds__(128) void attention_kernel(
    const unsigned short* __restrict__ Qb, const unsigned short* __restrict__ Kb,
    const unsigned short* __restrict__ Vt, unsigned short* __restrict__ Ob) {
  __shared__ float          sc_lds[4 * 8 * 32 * 8];   // [b][tile][lane][r] 32 KB
  __shared__ unsigned short at_lds[4 * 64 * LDP];     // [b][q][key] pitch 40, 20 KB

  const int tid  = threadIdx.x;
  const int lane = tid & 31;
  const int b    = tid >> 5;                // wave id == batch index
  const int h    = blockIdx.x;
  const int q0   = blockIdx.y * 64;
  const int hb   = h * 4 + b;
  const int l15  = lane & 15;
  const int lh   = lane >> 4;

  const unsigned short* Q = Qb + (size_t)hb * (S_LEN * HD);
  const unsigned short* K = Kb + (size_t)hb * (S_LEN * HD);
  const unsigned short* V = Vt + (size_t)hb * (S_LEN * HD);  // [d][key]
  unsigned short*       O = Ob + (size_t)hb * (S_LEN * HD);

  // Preload Q A-fragments for the whole q-tile (reused every k-step).
  BF16Frag qf[4][2];
#pragma unroll
  for (int mi = 0; mi < 4; ++mi)
#pragma unroll
    for (int kc = 0; kc < 2; ++kc) {
      const unsigned short* p =
          Q + (size_t)(q0 + mi * 16 + l15) * HD + kc * 32 + lh * 8;
      qf[mi][kc].u4[0] = *(const uint4*)(p);
      qf[mi][kc].u4[1] = *(const uint4*)(p + 16);
    }

  F32x8 oacc[4][4];
#pragma unroll
  for (int mi = 0; mi < 4; ++mi)
#pragma unroll
    for (int ni = 0; ni < 4; ++ni)
#pragma unroll
      for (int r = 0; r < 8; ++r) oacc[mi][ni].f[r] = 0.0f;

  for (int kt = 0; kt < S_LEN; kt += 32) {
    // K^T B-fragments straight from global (contiguous along d).
    BF16Frag kf[2][2];
#pragma unroll
    for (int ni = 0; ni < 2; ++ni)
#pragma unroll
      for (int kc = 0; kc < 2; ++kc) {
        const unsigned short* p =
            K + (size_t)(kt + ni * 16 + l15) * HD + kc * 32 + lh * 8;
        kf[ni][kc].u4[0] = *(const uint4*)(p);
        kf[ni][kc].u4[1] = *(const uint4*)(p + 16);
      }
    // Scores S_b = Q @ K^T (64x32).
    F32x8 s[4][2];
#pragma unroll
    for (int mi = 0; mi < 4; ++mi)
#pragma unroll
      for (int ni = 0; ni < 2; ++ni)
#pragma unroll
        for (int r = 0; r < 8; ++r) s[mi][ni].f[r] = 0.0f;
#pragma unroll
    for (int kc = 0; kc < 2; ++kc)
#pragma unroll
      for (int mi = 0; mi < 4; ++mi)
#pragma unroll
        for (int ni = 0; ni < 2; ++ni)
          s[mi][ni].v = __builtin_amdgcn_wmma_f32_16x16x32_bf16(
              false, qf[mi][kc].v, false, kf[ni][kc].v, (short)0, s[mi][ni].v,
              false, false);

    __syncthreads();  // all waves done reading sc_lds from previous step
#pragma unroll
    for (int mi = 0; mi < 4; ++mi)
#pragma unroll
      for (int ni = 0; ni < 2; ++ni) {
        int t = mi * 2 + ni;
        float* dst = &sc_lds[(((b * 8 + t) * 32) + lane) * 8];
        *(float4*)(dst)     = s[mi][ni].f4[0];
        *(float4*)(dst + 4) = s[mi][ni].f4[1];
      }
    __syncthreads();  // scores from all 4 batches visible

    // Elementwise softmax over batch (axis of size 4), write own attn as bf16.
#pragma unroll
    for (int mi = 0; mi < 4; ++mi)
#pragma unroll
      for (int ni = 0; ni < 2; ++ni) {
        int t = mi * 2 + ni;
        F32x8 sb[4];
#pragma unroll
        for (int bb = 0; bb < 4; ++bb) {
          const float* p = &sc_lds[(((bb * 8 + t) * 32) + lane) * 8];
          sb[bb].f4[0] = *(const float4*)(p);
          sb[bb].f4[1] = *(const float4*)(p + 4);
        }
#pragma unroll
        for (int r = 0; r < 8; ++r) {
          float mx = fmaxf(fmaxf(sb[0].f[r], sb[1].f[r]),
                           fmaxf(sb[2].f[r], sb[3].f[r]));
          float e0 = __expf(sb[0].f[r] - mx);
          float e1 = __expf(sb[1].f[r] - mx);
          float e2 = __expf(sb[2].f[r] - mx);
          float e3 = __expf(sb[3].f[r] - mx);
          float inv = 1.0f / (e0 + e1 + e2 + e3);
          float mine = (b == 0) ? e0 : (b == 1) ? e1 : (b == 2) ? e2 : e3;
          int q   = mi * 16 + r + 8 * lh;   // C-layout row
          int key = ni * 16 + l15;          // C-layout col
          at_lds[(b * 64 + q) * LDP + key] = f2bf(mine * inv);
        }
      }
    // No barrier: at_lds section [b] is written and read by wave b only;
    // same-wave LDS ops are in-order (DScnt).

    // O_b += attn_b @ V_b  (contraction over the 32 keys of this tile).
    BF16Frag af[4], vf[4];
#pragma unroll
    for (int mi = 0; mi < 4; ++mi) {
      const unsigned short* p = &at_lds[(b * 64 + mi * 16 + l15) * LDP + lh * 8];
      af[mi].u4[0] = *(const uint4*)(p);
      af[mi].u4[1] = *(const uint4*)(p + 16);
    }
#pragma unroll
    for (int ni = 0; ni < 4; ++ni) {  // Vt[d][key]: two 16B chunks per lane, 32B apart
      const unsigned short* p = V + (size_t)(ni * 16 + l15) * S_LEN + kt + lh * 8;
      vf[ni].u4[0] = *(const uint4*)(p);
      vf[ni].u4[1] = *(const uint4*)(p + 16);
    }
#pragma unroll
    for (int mi = 0; mi < 4; ++mi)
#pragma unroll
      for (int ni = 0; ni < 4; ++ni)
        oacc[mi][ni].v = __builtin_amdgcn_wmma_f32_16x16x32_bf16(
            false, af[mi].v, false, vf[ni].v, (short)0, oacc[mi][ni].v,
            false, false);
  }

  // Store O flat at ((h*4+b)*2048+q)*64+d  (== row-major input of out-proj).
#pragma unroll
  for (int mi = 0; mi < 4; ++mi)
#pragma unroll
    for (int ni = 0; ni < 4; ++ni) {
      int q = q0 + mi * 16 + lh * 8;
      int d = ni * 16 + l15;
#pragma unroll
      for (int r = 0; r < 8; ++r)
        O[(size_t)(q + r) * HD + d] = f2bf(oacc[mi][ni].f[r]);
    }
}

// ---------------------------------------------------------------------------
extern "C" void kernel_launch(void* const* d_in, const int* in_sizes, int n_in,
                              void* d_out, int out_size, void* d_ws,
                              size_t ws_size, hipStream_t stream) {
  const float* x  = (const float*)d_in[0];
  const float* y  = (const float*)d_in[1];
  const float* Wq = (const float*)d_in[2];
  const float* bq = (const float*)d_in[3];
  const float* Wk = (const float*)d_in[4];
  const float* bk = (const float*)d_in[5];
  const float* Wv = (const float*)d_in[6];
  const float* bv = (const float*)d_in[7];
  const float* Wo = (const float*)d_in[8];
  const float* bo = (const float*)d_in[9];
  float* out = (float*)d_out;

  // Workspace: 4 bf16 buffers of 8192x1024 (16 MB each, 64 MB total).
  const size_t BUF = 8ull * 1024 * 1024;  // elements (u16)
  unsigned short* Qb = (unsigned short*)d_ws;
  unsigned short* Kb = Qb + BUF;
  unsigned short* Vt = Kb + BUF;
  unsigned short* Ob = Vt + BUF;

  qkv_proj_kernel<<<dim3(GN / 256, GM / 128, 3), 256, 0, stream>>>(
      x, y, Wq, bq, Wk, bk, Wv, bv, Qb, Kb, Vt);
  attention_kernel<<<dim3(16, S_LEN / 64), 128, 0, stream>>>(Qb, Kb, Vt, Ob);
  out_proj_kernel<<<dim3(GN / 256, GM / 128), 256, 0, stream>>>(Ob, Wo, bo, out);
}
